// MultiHeadAttention_37263136260157
// MI455X (gfx1250) — matrix operand order repaired and checked by reading, not compiled, not run on hardware
//
#include <hip/hip_runtime.h>
#include <hip/hip_bf16.h>

typedef __attribute__((ext_vector_type(16))) _Float16 v16h;
typedef __attribute__((ext_vector_type(8)))  float    v8f;

#define D_MODEL 1024
#define NH      16
#define DK      64
#define BATCH   4
#define SEQ     2048
#define QWAVES  8           // waves per block in flash kernel
#define PSTRIDE 72          // padded LDS row stride (halves) -> conflict-free b128 reads

union HF16 { v16h v; _Float16 h[16]; uint4 u[2]; };

// Wave-local LDS ordering: per-wave LDS ops are in-order at the LDS unit, so a
// dscnt drain + compiler memory barrier is enough (each wave owns its slice).
__device__ inline void lds_wave_fence() {
  asm volatile("s_wait_dscnt 0" ::: "memory");
}

// ---- WMMA fragment loaders (layouts per CDNA5 ISA 7.12.2) -------------------

// A (16x32 f16): lane holds row M=lane&15; elems 0..7 = K kBase+8*hl+0..7,
// elems 8..15 = K kBase+8*hl+16..23  (hl = lane>=16)
__device__ inline v16h load_a_f16(const _Float16* A, int lda, int mBase, int kBase, int lane) {
  int m = mBase + (lane & 15);
  int k = kBase + ((lane & 16) >> 1);          // +8 for upper half-wave
  const _Float16* p = A + (size_t)m * lda + k;
  HF16 f;
  f.u[0] = *(const uint4*)(p);
  f.u[1] = *(const uint4*)(p + 16);
  return f.v;
}

__device__ inline v16h load_a_f32(const float* A, int lda, int mBase, int kBase, int lane) {
  int m = mBase + (lane & 15);
  int k = kBase + ((lane & 16) >> 1);
  const float* p = A + (size_t)m * lda + k;
  float4 x0 = ((const float4*)p)[0];
  float4 x1 = ((const float4*)p)[1];
  float4 y0 = ((const float4*)(p + 16))[0];
  float4 y1 = ((const float4*)(p + 16))[1];
  HF16 f;
  f.h[0]=(_Float16)x0.x; f.h[1]=(_Float16)x0.y; f.h[2]=(_Float16)x0.z; f.h[3]=(_Float16)x0.w;
  f.h[4]=(_Float16)x1.x; f.h[5]=(_Float16)x1.y; f.h[6]=(_Float16)x1.z; f.h[7]=(_Float16)x1.w;
  f.h[8]=(_Float16)y0.x; f.h[9]=(_Float16)y0.y; f.h[10]=(_Float16)y0.z; f.h[11]=(_Float16)y0.w;
  f.h[12]=(_Float16)y1.x; f.h[13]=(_Float16)y1.y; f.h[14]=(_Float16)y1.z; f.h[15]=(_Float16)y1.w;
  return f.v;
}

// B (32x16 f16): lane holds col N=lane&15; elems 0..7 = K kBase+8*hl+0..7,
// elems 8..15 = K kBase+8*hl+16..23 (same order as A).
// Bcm is stored column-major: Bcm[n][k], ld = ldb.
__device__ inline v16h load_b_f16(const _Float16* Bcm, int ldb, int nBase, int kBase, int lane) {
  int n = nBase + (lane & 15);
  int k = kBase + ((lane & 16) >> 1);          // +8 for upper half-wave
  const _Float16* p = Bcm + (size_t)n * ldb + k;
  HF16 f;
  f.u[0] = *(const uint4*)(p);
  f.u[1] = *(const uint4*)(p + 16);
  return f.v;
}

__device__ inline v16h load_b_f32(const float* Bcm, int ldb, int nBase, int kBase, int lane) {
  int n = nBase + (lane & 15);
  int k = kBase + ((lane & 16) >> 1);
  const float* p = Bcm + (size_t)n * ldb + k;
  float4 x0 = ((const float4*)p)[0];
  float4 x1 = ((const float4*)p)[1];
  float4 x2 = ((const float4*)(p + 16))[0];
  float4 x3 = ((const float4*)(p + 16))[1];
  HF16 f;
  f.h[0]=(_Float16)x0.x; f.h[1]=(_Float16)x0.y; f.h[2]=(_Float16)x0.z; f.h[3]=(_Float16)x0.w;
  f.h[4]=(_Float16)x1.x; f.h[5]=(_Float16)x1.y; f.h[6]=(_Float16)x1.z; f.h[7]=(_Float16)x1.w;
  f.h[8]=(_Float16)x2.x; f.h[9]=(_Float16)x2.y; f.h[10]=(_Float16)x2.z; f.h[11]=(_Float16)x2.w;
  f.h[12]=(_Float16)x3.x; f.h[13]=(_Float16)x3.y; f.h[14]=(_Float16)x3.z; f.h[15]=(_Float16)x3.w;
  return f.v;
}

__device__ inline v8f wmma32(v16h a, v16h b, v8f c) {
  return __builtin_amdgcn_wmma_f32_16x16x32_f16(false, a, false, b, (short)0, c, false, false);
}

// ---- weight pack: w[h][d][e] (f32) -> Wcm[n=h*64+e][d] (f16) ----------------
__global__ __launch_bounds__(256)
void mha_pack_w(const float* __restrict__ w, _Float16* __restrict__ out) {
  int idx = blockIdx.x * 256 + threadIdx.x;      // over 1024*1024
  int n = idx >> 10, d = idx & 1023;
  int hh = n >> 6, e = n & 63;
  out[idx] = (_Float16)w[((size_t)hh * D_MODEL + d) * DK + e];
}

// ---- generic WMMA GEMM: C(MxN) = A(MxK) * Bcm(N x K, column-major source) ---
// One wave computes a 16x64 C tile (A fragment reused across 4 B tiles).
// SMODE 0: f16 store as qh/kh[b][h][s][e];  1: f16 store transposed vt[b][h][e][s];
// SMODE 2: f32 row-major store (final output).
template<int A_F32, int B_F32, int SMODE>
__global__ __launch_bounds__(256)
void mha_gemm(const void* __restrict__ Aptr, const void* __restrict__ Bptr,
              void* __restrict__ Cptr, int M, int N, int K) {
  int lane = threadIdx.x & 31;
  int wid  = (blockIdx.x * 256 + (int)threadIdx.x) >> 5;
  int mTiles = M >> 4;
  int mT = wid % mTiles, nT = wid / mTiles;
  int mBase = mT << 4, nBase = nT << 6;

  v8f acc[4] = {};
  for (int kb = 0; kb < K; kb += 32) {
    v16h a;
    if constexpr (A_F32) a = load_a_f32((const float*)Aptr, K, mBase, kb, lane);
    else                 a = load_a_f16((const _Float16*)Aptr, K, mBase, kb, lane);
#pragma unroll
    for (int t = 0; t < 4; ++t) {
      v16h bfr;
      if constexpr (B_F32) bfr = load_b_f32((const float*)Bptr, K, nBase + t * 16, kb, lane);
      else                 bfr = load_b_f16((const _Float16*)Bptr, K, nBase + t * 16, kb, lane);
      acc[t] = wmma32(a, bfr, acc[t]);
    }
  }

  int row0 = (lane & 16) >> 1;     // 0 or 8
  int col  = lane & 15;
#pragma unroll
  for (int t = 0; t < 4; ++t) {
#pragma unroll
    for (int r = 0; r < 8; ++r) {
      int m = mBase + row0 + r;
      int n = nBase + t * 16 + col;
      if constexpr (SMODE == 2) {
        ((float*)Cptr)[(size_t)m * N + n] = acc[t][r];
      } else {
        int bb = m >> 11, s = m & (SEQ - 1);       // m = b*SEQ + s
        int hh = n >> 6,  e = n & 63;              // n = h*64 + e
        if constexpr (SMODE == 0)
          ((_Float16*)Cptr)[(((size_t)(bb * NH + hh)) * SEQ + s) * DK + e] = (_Float16)acc[t][r];
        else
          ((_Float16*)Cptr)[(((size_t)(bb * NH + hh)) * DK + e) * SEQ + s] = (_Float16)acc[t][r];
      }
    }
  }
}

// ---- flash attention: 1 wave = 16 queries, key tiles of 64 ------------------
__global__ __launch_bounds__(256)
void mha_flash(const _Float16* __restrict__ qh, const _Float16* __restrict__ kh,
               const _Float16* __restrict__ vt, const unsigned char* __restrict__ mask,
               _Float16* __restrict__ ctx) {
  __shared__ _Float16 pbuf[QWAVES][16 * PSTRIDE];
  int lane = threadIdx.x & 31;
  int wv   = threadIdx.x >> 5;
  int hh = blockIdx.y, bb = blockIdx.z;
  int qBase = (blockIdx.x * QWAVES + wv) << 4;
  size_t bh = (size_t)(bb * NH + hh);
  const _Float16* Q  = qh + bh * SEQ * DK;
  const _Float16* Kh = kh + bh * SEQ * DK;   // row-major [key][e] == B column-major
  const _Float16* Vt = vt + bh * DK * SEQ;   // [e][key]           == B column-major
  const unsigned char* mb = mask + (size_t)bb * SEQ;

  int col = lane & 15;
  int hl  = lane >> 4;
  _Float16* pb = pbuf[wv];

  v16h qa0 = load_a_f16(Q, DK, qBase, 0, lane);
  v16h qa1 = load_a_f16(Q, DK, qBase, 32, lane);

  float mrow[8], lrow[8];
  v8f o[4] = {};
#pragma unroll
  for (int r = 0; r < 8; ++r) { mrow[r] = -1e30f; lrow[r] = 0.f; }

  for (int kb = 0; kb < SEQ; kb += 64) {
    v8f z = {};
    // scores: 4 x (16x16) key tiles, contraction over e = 2 WMMAs each
    v8f s0 = wmma32(qa0, load_b_f16(Kh, DK, kb,       0, lane), z);
    s0     = wmma32(qa1, load_b_f16(Kh, DK, kb,      32, lane), s0);
    v8f s1 = wmma32(qa0, load_b_f16(Kh, DK, kb + 16,  0, lane), z);
    s1     = wmma32(qa1, load_b_f16(Kh, DK, kb + 16, 32, lane), s1);
    v8f s2 = wmma32(qa0, load_b_f16(Kh, DK, kb + 32,  0, lane), z);
    s2     = wmma32(qa1, load_b_f16(Kh, DK, kb + 32, 32, lane), s2);
    v8f s3 = wmma32(qa0, load_b_f16(Kh, DK, kb + 48,  0, lane), z);
    s3     = wmma32(qa1, load_b_f16(Kh, DK, kb + 48, 32, lane), s3);

    float bias0 = mb[kb + col]      ? 0.f : -1e30f;
    float bias1 = mb[kb + 16 + col] ? 0.f : -1e30f;
    float bias2 = mb[kb + 32 + col] ? 0.f : -1e30f;
    float bias3 = mb[kb + 48 + col] ? 0.f : -1e30f;

    float al[8];
#pragma unroll
    for (int r = 0; r < 8; ++r) {
      float v0 = s0[r] * 0.125f + bias0;     // 1/sqrt(64)
      float v1 = s1[r] * 0.125f + bias1;
      float v2 = s2[r] * 0.125f + bias2;
      float v3 = s3[r] * 0.125f + bias3;
      float tm = fmaxf(fmaxf(v0, v1), fmaxf(v2, v3));
      tm = fmaxf(tm, __shfl_xor(tm, 1, 32));
      tm = fmaxf(tm, __shfl_xor(tm, 2, 32));
      tm = fmaxf(tm, __shfl_xor(tm, 4, 32));
      tm = fmaxf(tm, __shfl_xor(tm, 8, 32));
      float nm = fmaxf(mrow[r], tm);
      al[r] = __expf(mrow[r] - nm);
      v0 = __expf(v0 - nm);
      v1 = __expf(v1 - nm);
      v2 = __expf(v2 - nm);
      v3 = __expf(v3 - nm);
      float rs = (v0 + v1) + (v2 + v3);
      rs += __shfl_xor(rs, 1, 32);
      rs += __shfl_xor(rs, 2, 32);
      rs += __shfl_xor(rs, 4, 32);
      rs += __shfl_xor(rs, 8, 32);
      lrow[r] = lrow[r] * al[r] + rs;
      mrow[r] = nm;
      s0[r] = v0; s1[r] = v1; s2[r] = v2; s3[r] = v3;
    }
#pragma unroll
    for (int t = 0; t < 4; ++t)
#pragma unroll
      for (int r = 0; r < 8; ++r) o[t][r] *= al[r];

    // P (16x64, C layout) -> private LDS slice (row-major) -> two A fragments
    lds_wave_fence();                         // protect prior reads (wave-local)
#pragma unroll
    for (int r = 0; r < 8; ++r) {
      int row = r + (hl << 3);
      pb[row * PSTRIDE + col]      = (_Float16)s0[r];
      pb[row * PSTRIDE + 16 + col] = (_Float16)s1[r];
      pb[row * PSTRIDE + 32 + col] = (_Float16)s2[r];
      pb[row * PSTRIDE + 48 + col] = (_Float16)s3[r];
    }
    lds_wave_fence();                         // stores visible before reads
    v16h pa0 = load_a_f16(pb, PSTRIDE, 0, 0,  lane);
    v16h pa1 = load_a_f16(pb, PSTRIDE, 0, 32, lane);

#pragma unroll
    for (int t = 0; t < 4; ++t) {
      v16h vb0 = load_b_f16(Vt, SEQ, t * 16, kb,      lane);
      v16h vb1 = load_b_f16(Vt, SEQ, t * 16, kb + 32, lane);
      o[t] = wmma32(pa0, vb0, o[t]);
      o[t] = wmma32(pa1, vb1, o[t]);
    }
  }

  float inv[8];
#pragma unroll
  for (int r = 0; r < 8; ++r) inv[r] = 1.0f / lrow[r];
#pragma unroll
  for (int t = 0; t < 4; ++t)
#pragma unroll
    for (int r = 0; r < 8; ++r) {
      int q = qBase + r + (hl << 3);
      int n = hh * DK + t * 16 + col;
      ctx[(size_t)(bb * SEQ + q) * D_MODEL + n] = (_Float16)(o[t][r] * inv[r]);
    }
}

// ---- host launch ------------------------------------------------------------
extern "C" void kernel_launch(void* const* d_in, const int* in_sizes, int n_in,
                              void* d_out, int out_size, void* d_ws, size_t ws_size,
                              hipStream_t stream) {
  const float* q  = (const float*)d_in[0];
  const float* k  = (const float*)d_in[1];
  const float* v  = (const float*)d_in[2];
  const unsigned char* mask = (const unsigned char*)d_in[3];
  const float* w_q = (const float*)d_in[4];
  const float* w_k = (const float*)d_in[5];
  const float* w_v = (const float*)d_in[6];
  const float* w_o = (const float*)d_in[7];
  float* out = (float*)d_out;

  char* ws = (char*)d_ws;
  size_t off = 0;
  auto alloc = [&](size_t bytes) -> void* {
    void* p = ws + off; off += (bytes + 255) & ~(size_t)255; return p;
  };
  const size_t WB = (size_t)D_MODEL * D_MODEL * sizeof(_Float16);              // 2 MB
  const size_t HB = (size_t)BATCH * NH * SEQ * DK * sizeof(_Float16);          // 16 MB
  _Float16* wq_cm = (_Float16*)alloc(WB);
  _Float16* wk_cm = (_Float16*)alloc(WB);
  _Float16* wv_cm = (_Float16*)alloc(WB);
  _Float16* qh  = (_Float16*)alloc(HB);
  _Float16* kh  = (_Float16*)alloc(HB);
  _Float16* vt  = (_Float16*)alloc(HB);
  _Float16* ctx = (_Float16*)alloc(HB);
  (void)in_sizes; (void)n_in; (void)out_size; (void)ws_size;

  // 1) pack QKV weights to f16 column-major
  mha_pack_w<<<(D_MODEL * D_MODEL) / 256, 256, 0, stream>>>(w_q, wq_cm);
  mha_pack_w<<<(D_MODEL * D_MODEL) / 256, 256, 0, stream>>>(w_k, wk_cm);
  mha_pack_w<<<(D_MODEL * D_MODEL) / 256, 256, 0, stream>>>(w_v, wv_cm);

  // 2) projections (M=8192, N=1024, K=1024); 16x64 tile per wave, 8 waves/block
  const int M = BATCH * SEQ;
  const int gemmBlocks = (M / 16) * (D_MODEL / 64) / 8;   // 1024
  mha_gemm<1, 0, 0><<<gemmBlocks, 256, 0, stream>>>(q, wq_cm, qh, M, D_MODEL, D_MODEL);
  mha_gemm<1, 0, 0><<<gemmBlocks, 256, 0, stream>>>(k, wk_cm, kh, M, D_MODEL, D_MODEL);
  mha_gemm<1, 0, 1><<<gemmBlocks, 256, 0, stream>>>(v, wv_cm, vt, M, D_MODEL, D_MODEL);

  // 3) flash attention
  mha_flash<<<dim3(SEQ / 16 / QWAVES, NH, BATCH), 256, 0, stream>>>(qh, kh, vt, mask, ctx);

  // 4) output projection: out = ctx @ w_o^T  (w_o row-major is already B-col-major)
  mha_gemm<0, 1, 2><<<gemmBlocks, 256, 0, stream>>>(ctx, w_o, out, M, D_MODEL, D_MODEL);
}
